// BERTRewardModel_74242804678835
// MI455X (gfx1250) — matrix operand order repaired and checked by reading, not compiled, not run on hardware
//
#include <hip/hip_runtime.h>
#include <math.h>
#include <stdint.h>

// ---------------- model constants ----------------
#define LNUM 22
#define DIM  768
#define NH   12
#define HDIM 64
#define IDIM 1152
#define BB   8
#define SS   1024
#define TOK  (BB*SS)           // 8192 tokens
#define EPSV 1e-5f

// ---------------- WMMA / vector types ----------------
typedef __bf16 v16bf __attribute__((ext_vector_type(16)));
typedef float  v8f   __attribute__((ext_vector_type(8)));
typedef unsigned short u16x8  __attribute__((ext_vector_type(8)));
typedef unsigned short u16x16 __attribute__((ext_vector_type(16)));
typedef int v4i __attribute__((ext_vector_type(4)));

__device__ __forceinline__ unsigned short f2bf(float f) {
  unsigned int u = __float_as_uint(f);
  u += 0x7FFFu + ((u >> 16) & 1u);     // round-to-nearest-even
  return (unsigned short)(u >> 16);
}

// ---------------- async global->LDS copy (gfx1250), guarded ----------------
#if defined(__has_builtin)
#if __has_builtin(__builtin_amdgcn_global_load_async_to_lds_b128)
#define HAVE_ASYNC_B128 1
#endif
#endif
#ifndef HAVE_ASYNC_B128
#define HAVE_ASYNC_B128 0
#endif

__device__ __forceinline__ void cp_async16(void* lds, const void* g) {
#if HAVE_ASYNC_B128
  __builtin_amdgcn_global_load_async_to_lds_b128(
      (__attribute__((address_space(1))) v4i*)(void*)g,
      (__attribute__((address_space(3))) v4i*)lds,
      0, 0);
#else
  *(uint4*)lds = *(const uint4*)g;
#endif
}

template<int Ncnt>
__device__ __forceinline__ void async_wait() {
#if HAVE_ASYNC_B128
#if __has_builtin(__builtin_amdgcn_s_wait_asynccnt)
  __builtin_amdgcn_s_wait_asynccnt(Ncnt);
#else
  asm volatile("s_wait_asynccnt %0" :: "i"(Ncnt) : "memory");
#endif
#endif
}

// ---------------- weight transform: fp32 (K,N) -> bf16 (N,K) ----------------
__global__ __launch_bounds__(256)
void wtrans_kernel(const float* __restrict__ in, unsigned short* __restrict__ out,
                   int K, int N) {
  __shared__ float tile[32][33];
  int k0 = blockIdx.y * 32, n0 = blockIdx.x * 32;
  int c = threadIdx.x & 31, r8 = threadIdx.x >> 5;     // 8 rows per pass
#pragma unroll
  for (int rr = 0; rr < 32; rr += 8)
    tile[r8 + rr][c] = in[(size_t)(k0 + r8 + rr) * N + (n0 + c)];
  __syncthreads();
#pragma unroll
  for (int rr = 0; rr < 32; rr += 8) {
    int n = r8 + rr;
    out[(size_t)(n0 + n) * K + (k0 + c)] = f2bf(tile[c][n]);
  }
}

// ---------------- GEMM: C[M,N](f32) (+)= A[M,K](bf16) x Bt[N,K](bf16) -------
// block 256 = 8 waves; tile 128x128, BK=32; wave = 64 rows x 32 cols (8 WMMA)
// double-buffered LDS, async-copy pipelined; last iteration peeled.
#define ASTR 40
#define BSTR 40

template<int N, int K, bool RES>
__global__ __launch_bounds__(256)
void gemm_bf16(const unsigned short* __restrict__ A,
               const unsigned short* __restrict__ Bt,
               float* __restrict__ C) {
  __shared__ __align__(16) unsigned short As[2][128 * ASTR];
  __shared__ __align__(16) unsigned short Bs[2][128 * BSTR];

  const int tid  = threadIdx.x;
  const int lane = tid & 31;
  const int w    = tid >> 5;
  const int wm   = w & 1;       // 2 wave-rows of 64
  const int wn   = w >> 1;      // 4 wave-cols of 32
  const int bm0  = blockIdx.y * 128;
  const int bn0  = blockIdx.x * 128;

  auto stage = [&](int buf, int k0) {
#pragma unroll
    for (int i = 0; i < 2; ++i) {
      int id = tid + i * 256;
      int r  = id >> 2;
      int c4 = (id & 3) * 8;
      cp_async16(&As[buf][r * ASTR + c4], A  + (size_t)(bm0 + r) * K + k0 + c4);
      cp_async16(&Bs[buf][r * BSTR + c4], Bt + (size_t)(bn0 + r) * K + k0 + c4);
    }
  };

  const v8f zero = {0.f,0.f,0.f,0.f,0.f,0.f,0.f,0.f};
  v8f acc[4][2];
#pragma unroll
  for (int tm = 0; tm < 4; ++tm) { acc[tm][0] = zero; acc[tm][1] = zero; }

  auto compute = [&](int cur) {
    v16bf af[4], bfr[2];
#pragma unroll
    for (int tm = 0; tm < 4; ++tm) {
      int row = wm * 64 + tm * 16 + (lane & 15);
      int kb  = (lane < 16) ? 0 : 8;          // A 16x32 bf16 lane layout
      u16x8 lo = *(const u16x8*)(&As[cur][row * ASTR + kb]);
      u16x8 hi = *(const u16x8*)(&As[cur][row * ASTR + kb + 16]);
      u16x16 t;
#pragma unroll
      for (int i = 0; i < 8; ++i) { t[i] = lo[i]; t[i + 8] = hi[i]; }
      af[tm] = __builtin_bit_cast(v16bf, t);
    }
#pragma unroll
    for (int tn = 0; tn < 2; ++tn) {
      int col = wn * 32 + tn * 16 + (lane & 15);
      int ks  = (lane < 16) ? 0 : 8;          // B 32x16 bf16 lane layout
      u16x8 lo = *(const u16x8*)(&Bs[cur][col * BSTR + ks]);
      u16x8 hi = *(const u16x8*)(&Bs[cur][col * BSTR + ks + 16]);
      u16x16 t;
#pragma unroll
      for (int i = 0; i < 8; ++i) { t[i] = lo[i]; t[i + 8] = hi[i]; }
      bfr[tn] = __builtin_bit_cast(v16bf, t);
    }
#pragma unroll
    for (int tm = 0; tm < 4; ++tm)
#pragma unroll
      for (int tn = 0; tn < 2; ++tn)
        acc[tm][tn] = __builtin_amdgcn_wmma_f32_16x16x32_bf16(
            false, af[tm], false, bfr[tn], (short)0, acc[tm][tn], false, false);
  };

  stage(0, 0);                                // prologue: tile 0 in flight (4 ops/wave)

  int k0 = 0;
  for (; k0 < K - 32; k0 += 32) {             // steady state: branchless wait<4>
    const int cur = (k0 >> 5) & 1;
    stage(cur ^ 1, k0 + 32);                  // issue next tile (<=8 in flight)
    async_wait<4>();                          // oldest 4 (current tile) complete
    __syncthreads();
    compute(cur);
    __syncthreads();                          // all waves done reading 'cur'
  }
  {                                           // peeled last tile
    const int cur = (k0 >> 5) & 1;
    async_wait<0>();
    __syncthreads();
    compute(cur);
  }

  // epilogue: C layout lane<16 -> N=lane,M=r ; lane>=16 -> N=lane-16,M=8+r
#pragma unroll
  for (int tm = 0; tm < 4; ++tm)
#pragma unroll
    for (int tn = 0; tn < 2; ++tn) {
      int col   = bn0 + wn * 32 + tn * 16 + (lane & 15);
      int rbase = bm0 + wm * 64 + tm * 16 + ((lane < 16) ? 0 : 8);
      float* cp = C + (size_t)rbase * N + col;
#pragma unroll
      for (int r = 0; r < 8; ++r) {
        if (RES) cp[(size_t)r * N] += acc[tm][tn][r];
        else     cp[(size_t)r * N]  = acc[tm][tn][r];
      }
    }
}

// ---------------- LayerNorm ----------------
__device__ __forceinline__ void ln_stats(const float* row, float* red,
                                         float& mean, float& rstd) {
  float s = 0.f;
  for (int i = threadIdx.x; i < DIM; i += 256) s += row[i];
  red[threadIdx.x] = s; __syncthreads();
  for (int o = 128; o > 0; o >>= 1) {
    if (threadIdx.x < o) red[threadIdx.x] += red[threadIdx.x + o];
    __syncthreads();
  }
  mean = red[0] / DIM;
  __syncthreads();
  float vs = 0.f;
  for (int i = threadIdx.x; i < DIM; i += 256) { float d = row[i] - mean; vs += d * d; }
  red[threadIdx.x] = vs; __syncthreads();
  for (int o = 128; o > 0; o >>= 1) {
    if (threadIdx.x < o) red[threadIdx.x] += red[threadIdx.x + o];
    __syncthreads();
  }
  rstd = rsqrtf(red[0] / DIM + EPSV);
}

__global__ __launch_bounds__(256)
void ln_bf16_kernel(const float* __restrict__ in, const float* __restrict__ wgt,
                    unsigned short* __restrict__ out) {
  __shared__ float red[256];
  size_t t = blockIdx.x;
  const float* row = in + t * DIM;
  float mean, rstd;
  ln_stats(row, red, mean, rstd);
  for (int i = threadIdx.x; i < DIM; i += 256)
    out[t * DIM + i] = f2bf((row[i] - mean) * rstd * wgt[i]);
}

__global__ __launch_bounds__(256)
void ln_f32_kernel(const float* __restrict__ in, const float* __restrict__ wgt,
                   float* __restrict__ out) {
  __shared__ float red[256];
  size_t t = blockIdx.x;
  const float* row = in + t * DIM;
  float mean, rstd;
  ln_stats(row, red, mean, rstd);
  for (int i = threadIdx.x; i < DIM; i += 256)
    out[t * DIM + i] = (row[i] - mean) * rstd * wgt[i];
}

// embed gather + LN -> x (fp32 residual stream) and xbf (bf16 GEMM input)
__global__ __launch_bounds__(256)
void embed_ln_kernel(const int* __restrict__ ids, const float* __restrict__ emb,
                     const float* __restrict__ wgt, float* __restrict__ x,
                     unsigned short* __restrict__ xbf) {
  __shared__ float red[256];
  size_t t = blockIdx.x;
  const float* row = emb + (size_t)ids[t] * DIM;
  float mean, rstd;
  ln_stats(row, red, mean, rstd);
  for (int i = threadIdx.x; i < DIM; i += 256) {
    float v = (row[i] - mean) * rstd * wgt[i];
    x[t * DIM + i] = v;
    xbf[t * DIM + i] = f2bf(v);
  }
}

// ---------------- RoPE + split qkv -> (B,H,S,HD) fp32 ----------------
__global__ __launch_bounds__(256)
void rope_split(const float* __restrict__ qkv, float* __restrict__ q,
                float* __restrict__ k, float* __restrict__ v, float theta) {
  int t = blockIdx.x;
  int b = t / SS, s = t - b * SS;
  float lt = logf(theta);
  for (int c = threadIdx.x; c < 3 * DIM; c += 256) {
    int which = c / DIM;
    int rem   = c - which * DIM;
    int hh    = rem >> 6, d = rem & 63;
    float val = qkv[(size_t)t * (3 * DIM) + c];
    size_t oidx = ((size_t)(b * NH + hh) * SS + s) * HDIM + d;
    if (which == 2) { v[oidx] = val; continue; }
    int i = d & 31;
    float inv = expf(-((float)i / 32.f) * lt);
    float f = (float)s * inv;
    float cs = cosf(f), sn = sinf(f);
    float pair = qkv[(size_t)t * (3 * DIM) + which * DIM + hh * 64 + ((d < 32) ? (d + 32) : (d - 32))];
    float rot = (d < 32) ? -pair : pair;
    float res = val * cs + rot * sn;
    if (which == 0) q[oidx] = res; else k[oidx] = res;
  }
}

// ---------------- streaming-softmax attention (bf16 output) ----------------
// double-buffered async K/V staging; 32 async ops/wave per tile; tail peeled.
__global__ __launch_bounds__(64)
void attn_kernel(const float* __restrict__ q, const float* __restrict__ k,
                 const float* __restrict__ v, const int* __restrict__ amask,
                 unsigned short* __restrict__ out, int is_global) {
  __shared__ __align__(16) float Kt[2][64][64];
  __shared__ __align__(16) float Vt[2][64][64];
  __shared__ float Sc[64][64];
  __shared__ int pm[2][64];

  int bh = blockIdx.y;
  int b = bh / NH, hh = bh - b * NH;
  int s0 = blockIdx.x * 64;
  int tid = threadIdx.x;
  int sq = s0 + tid;

  auto stage = [&](int buf, int kt) {
    int kr = kt + tid;
    const float* krow = k + ((size_t)bh * SS + kr) * HDIM;
    const float* vrow = v + ((size_t)bh * SS + kr) * HDIM;
#pragma unroll
    for (int j = 0; j < 16; ++j) {
      cp_async16(&Kt[buf][tid][j * 4], krow + j * 4);
      cp_async16(&Vt[buf][tid][j * 4], vrow + j * 4);
    }
    pm[buf][tid] = amask[b * SS + kr];
  };

  float qv[64], acc[64];
  const float* qrow = q + ((size_t)bh * SS + sq) * HDIM;
#pragma unroll
  for (int d = 0; d < 64; ++d) { qv[d] = qrow[d]; acc[d] = 0.f; }
  float m = -3.0e38f, ssum = 0.f;

  int lo = is_global ? 0 : max(0, s0 - 64);
  int hi = is_global ? SS : min(SS, s0 + 128);

  auto tile_compute = [&](int cur, int kt) {
    float tm = -3.0e38f;
    for (int kk = 0; kk < 64; ++kk) {
      int kidx = kt + kk;
      int dd = sq - kidx;
      bool ok = (pm[cur][kk] != 0) && (is_global || (dd <= 64 && dd >= -64));
      float dsum = 0.f;
#pragma unroll
      for (int d = 0; d < 64; ++d) dsum += qv[d] * Kt[cur][kk][d];
      float scv = ok ? dsum * 0.125f : -3.0e38f;
      Sc[kk][tid] = scv;
      tm = fmaxf(tm, scv);
    }
    if (tm > -1.0e38f) {
      float nm = fmaxf(m, tm);
      float corr = __expf(m - nm);
      ssum *= corr;
#pragma unroll
      for (int d = 0; d < 64; ++d) acc[d] *= corr;
      for (int kk = 0; kk < 64; ++kk) {
        float scv = Sc[kk][tid];
        if (scv > -1.0e38f) {
          float p = __expf(scv - nm);
          ssum += p;
#pragma unroll
          for (int d = 0; d < 64; ++d) acc[d] += p * Vt[cur][kk][d];
        }
      }
      m = nm;
    }
  };

  stage(0, lo);
  int kt = lo;
  for (; kt < hi - 64; kt += 64) {            // steady state: wait<32>
    const int cur = ((kt - lo) >> 6) & 1;
    stage(cur ^ 1, kt + 64);
    async_wait<32>();
    __syncthreads();
    tile_compute(cur, kt);
    __syncthreads();
  }
  {                                           // peeled last tile
    const int cur = ((kt - lo) >> 6) & 1;
    async_wait<0>();
    __syncthreads();
    tile_compute(cur, kt);
  }

  float inv = (ssum > 0.f) ? (1.f / ssum) : 0.f;
  unsigned short* orow = out + (size_t)(b * SS + sq) * DIM + hh * HDIM;
#pragma unroll
  for (int d = 0; d < 64; ++d) orow[d] = f2bf(acc[d] * inv);
}

// ---------------- gated GELU (bf16 output) ----------------
__global__ __launch_bounds__(256)
void gelugate_kernel(const float* __restrict__ a, unsigned short* __restrict__ g, int n) {
  int idx = blockIdx.x * 256 + threadIdx.x;
  if (idx >= n) return;
  int t = idx / IDIM, i = idx - t * IDIM;
  float xa = a[(size_t)t * (2 * IDIM) + i];
  float gt = a[(size_t)t * (2 * IDIM) + IDIM + i];
  float ge = 0.5f * xa * (1.f + erff(xa * 0.70710678118654752f));
  g[idx] = f2bf(ge * gt);
}

// ---------------- pooling + classifier ----------------
__global__ __launch_bounds__(256)
void pool_kernel(const float* __restrict__ xin, const int* __restrict__ amask,
                 float* __restrict__ pooled) {
  int b = blockIdx.x;
  __shared__ float cnt_s;
  if (threadIdx.x == 0) {
    int c = 0;
    for (int s = 0; s < SS; ++s) c += amask[b * SS + s];
    cnt_s = fmaxf((float)c, 1e-9f);
  }
  __syncthreads();
  float inv = 1.f / cnt_s;
  for (int d = threadIdx.x; d < DIM; d += 256) {
    float s = 0.f;
    for (int t = 0; t < SS; ++t)
      if (amask[b * SS + t]) s += xin[(size_t)(b * SS + t) * DIM + d];
    pooled[b * DIM + d] = s * inv;
  }
}

__global__ __launch_bounds__(256)
void cls_kernel(const float* __restrict__ pooled, const float* __restrict__ cw,
                const float* __restrict__ cb, float* __restrict__ outp) {
  int b = blockIdx.x;
  __shared__ float red[256];
  float s = 0.f;
  for (int d = threadIdx.x; d < DIM; d += 256) s += pooled[b * DIM + d] * cw[d];
  red[threadIdx.x] = s; __syncthreads();
  for (int o = 128; o > 0; o >>= 1) {
    if (threadIdx.x < o) red[threadIdx.x] += red[threadIdx.x + o];
    __syncthreads();
  }
  if (threadIdx.x == 0) {
    float logit = red[0] + cb[0];
    outp[b]      = 1.f / (1.f + __expf(-logit));   // scores_out
    outp[BB + b] = logit;                          // logits
  }
}

// ---------------- orchestration ----------------
extern "C" void kernel_launch(void* const* d_in, const int* in_sizes, int n_in,
                              void* d_out, int out_size, void* d_ws, size_t ws_size,
                              hipStream_t stream) {
  (void)in_sizes; (void)n_in; (void)out_size; (void)ws_size;
  const int*   input_ids    = (const int*)  d_in[0];
  const int*   amask        = (const int*)  d_in[1];
  const float* tok_emb      = (const float*)d_in[2];
  const float* emb_norm_w   = (const float*)d_in[3];
  const float* attn_norm_w  = (const float*)d_in[4];
  const float* Wqkv         = (const float*)d_in[5];
  const float* Wo           = (const float*)d_in[6];
  const float* mlp_norm_w   = (const float*)d_in[7];
  const float* Wi           = (const float*)d_in[8];
  const float* Wmo          = (const float*)d_in[9];
  const float* final_norm_w = (const float*)d_in[10];
  const float* cls_w        = (const float*)d_in[11];
  const float* cls_b        = (const float*)d_in[12];
  float* out = (float*)d_out;

  const size_t XSZ   = (size_t)TOK * DIM;        // 6,291,456
  const size_t QKVSZ = (size_t)TOK * 3 * DIM;    // 18,874,368
  char* base = (char*)d_ws;
  float* x      = (float*)base;                          base += XSZ * 4;
  float* qkv    = (float*)base;                          base += QKVSZ * 4;   // also MLP 'a', also final-LN out
  float* qb     = (float*)base;                          base += XSZ * 4;
  float* kb     = (float*)base;                          base += XSZ * 4;
  float* vb     = (float*)base;                          base += XSZ * 4;
  float* pooled = (float*)base;                          base += (size_t)BB * DIM * 4;
  unsigned short* xbf = (unsigned short*)base;           base += XSZ * 2;
  unsigned short* hbf = (unsigned short*)base;           base += XSZ * 2;
  unsigned short* abf = (unsigned short*)base;           base += (size_t)TOK * IDIM * 2;  // attn out / gelu out
  unsigned short* wtb = (unsigned short*)base;           base += (size_t)DIM * 3 * DIM * 2; // transposed bf16 weight panel

  embed_ln_kernel<<<TOK, 256, 0, stream>>>(input_ids, tok_emb, emb_norm_w, x, xbf);

  for (int l = 0; l < LNUM; ++l) {
    int is_global = (l % 3 == 0) ? 1 : 0;

    const unsigned short* Abf = xbf;
    if (l != 0) {
      ln_bf16_kernel<<<TOK, 256, 0, stream>>>(x, attn_norm_w + (size_t)l * DIM, hbf);
      Abf = hbf;
    }

    // qkv = h @ Wqkv[l]   (8192 x 768 x 2304)
    wtrans_kernel<<<dim3(3 * DIM / 32, DIM / 32), 256, 0, stream>>>(
        Wqkv + (size_t)l * DIM * 3 * DIM, wtb, DIM, 3 * DIM);
    gemm_bf16<3 * DIM, DIM, false><<<dim3(3 * DIM / 128, TOK / 128), 256, 0, stream>>>(
        Abf, wtb, qkv);

    rope_split<<<TOK, 256, 0, stream>>>(qkv, qb, kb, vb,
                                        is_global ? 160000.f : 10000.f);

    attn_kernel<<<dim3(SS / 64, BB * NH), 64, 0, stream>>>(
        qb, kb, vb, amask, abf, is_global);

    // x += attn @ Wo[l]   (8192 x 768 x 768), residual fused
    wtrans_kernel<<<dim3(DIM / 32, DIM / 32), 256, 0, stream>>>(
        Wo + (size_t)l * DIM * DIM, wtb, DIM, DIM);
    gemm_bf16<DIM, DIM, true><<<dim3(DIM / 128, TOK / 128), 256, 0, stream>>>(
        abf, wtb, x);

    ln_bf16_kernel<<<TOK, 256, 0, stream>>>(x, mlp_norm_w + (size_t)l * DIM, hbf);

    // a = hm @ Wi[l]      (8192 x 768 x 2304)
    wtrans_kernel<<<dim3(2 * IDIM / 32, DIM / 32), 256, 0, stream>>>(
        Wi + (size_t)l * DIM * 2 * IDIM, wtb, DIM, 2 * IDIM);
    gemm_bf16<2 * IDIM, DIM, false><<<dim3(2 * IDIM / 128, TOK / 128), 256, 0, stream>>>(
        hbf, wtb, qkv);

    gelugate_kernel<<<((TOK * IDIM) + 255) / 256, 256, 0, stream>>>(
        qkv, abf, TOK * IDIM);

    // x += g @ Wmo[l]     (8192 x 1152 x 768), residual fused
    wtrans_kernel<<<dim3(DIM / 32, IDIM / 32), 256, 0, stream>>>(
        Wmo + (size_t)l * IDIM * DIM, wtb, IDIM, DIM);
    gemm_bf16<DIM, IDIM, true><<<dim3(DIM / 128, TOK / 128), 256, 0, stream>>>(
        abf, wtb, x);
  }

  ln_f32_kernel<<<TOK, 256, 0, stream>>>(x, final_norm_w, qkv);
  pool_kernel<<<BB, 256, 0, stream>>>(qkv, amask, pooled);
  cls_kernel<<<BB, 256, 0, stream>>>(pooled, cls_w, cls_b, out);
}
